// MultiHeadAttention_34067680592087
// MI455X (gfx1250) — matrix operand order repaired and checked by reading, not compiled, not run on hardware
//
#include <hip/hip_runtime.h>
#include <hip/hip_bf16.h>

// ---------------------------------------------------------------------------
// MHA forward for MI455X (gfx1250): bf16 WMMA everywhere, flash-attention
// single pass, async-to-LDS double-buffered B tiles in the GEMMs.
// ---------------------------------------------------------------------------

typedef __attribute__((ext_vector_type(16))) __bf16 v16bf;
typedef __attribute__((ext_vector_type(8)))  float  v8f;

#define D_MODEL 768
#define NHEAD   12
#define HD      64
#define SEQ     2048
#define BATCH   4
#define MROWS   (BATCH * SEQ)   // 8192

static __device__ __forceinline__ v8f wmma_bf16(v16bf a, v16bf b, v8f c) {
  // D = A(16x32 bf16) * B(32x16 bf16) + C(16x16 f32)
  return __builtin_amdgcn_wmma_f32_16x16x32_bf16(
      /*neg_a=*/false, a, /*neg_b=*/false, b,
      /*c_mod=*/(short)0, c, /*reuse_a=*/false, /*reuse_b=*/false);
}

// A-fragment: 16x32 bf16, row-major source A[m][k], lda elements per row.
// k order used here (contiguous, matches B): lane%16 = M; lanes<16 hold K=0..15, lanes>=16 hold K=16..31.
static __device__ __forceinline__ v16bf
load_a16x32(const __bf16* __restrict A, int lda, int m_base, int k0, int lane) {
  int m  = m_base + (lane & 15);
  int kb = (lane < 16) ? 0 : 16;
  const __bf16* p = A + (long)m * lda + k0 + kb;
  v16bf a;
#pragma unroll
  for (int e = 0; e < 8; ++e)  a[e] = p[e];        // K = kb + e
#pragma unroll
  for (int e = 8; e < 16; ++e) a[e] = p[e];        // K = kb + e
  return a;
}

// B-fragment: 32x16 bf16 (KxN) from a TRANSPOSED row-major array Bt[n][k]
// (ldb elements per row): each lane reads 16 contiguous bf16.
// k order used here (contiguous, matches A): lane%16 = N; lanes<16 hold K=0..15, lanes>=16 hold K=16..31.
static __device__ __forceinline__ v16bf
load_bT32x16(const __bf16* __restrict Bt, long ldb, int n_base, int k0, int lane) {
  int n  = n_base + (lane & 15);
  int kb = (lane < 16) ? 0 : 16;
  const __bf16* p = Bt + (long)n * ldb + k0 + kb;
  v16bf b;
#pragma unroll
  for (int e = 0; e < 16; ++e) b[e] = p[e];
  return b;
}

// ---------------------------------------------------------------------------
// Async stage of one B tile (64 n-rows x 32 k) into LDS: 256 threads move
// 16B each via the CDNA5 async-copy path (tracked with ASYNCcnt).
// LDS layout: dst[n*32 + k], matching load_bT32x16 with ldb=32.
// ---------------------------------------------------------------------------
static __device__ __forceinline__ void
stage_b_async(const __bf16* __restrict Wt, int n0, int k0, int tid, __bf16* dst_lds) {
  int n  = tid >> 2;           // 0..63
  int kk = (tid & 3) * 8;      // 0,8,16,24
  const __bf16* g = Wt + (long)(n0 + n) * D_MODEL + k0 + kk;
  unsigned lds = (unsigned)(unsigned long long)(uintptr_t)(dst_lds + n * 32 + kk);
  unsigned long long ga = (unsigned long long)(uintptr_t)g;
  asm volatile("global_load_async_to_lds_b128 %0, %1, off"
               :: "v"(lds), "v"(ga) : "memory");
}
static __device__ __forceinline__ void wait_async_and_barrier() {
  asm volatile("s_wait_asynccnt 0x0" ::: "memory");
  __syncthreads();
}

// ---------------------------------------------------------------------------
// Precision prep: fp32 -> bf16 (x) and fp32 -> bf16 transposed (weights).
// ---------------------------------------------------------------------------
__global__ void cvt_x_kernel(const float* __restrict x, __bf16* __restrict xb) {
  int i = blockIdx.x * 256 + threadIdx.x;        // grid sized exactly
  xb[i] = (__bf16)x[i];
}

__global__ void cvt_w_kernel(const float* __restrict Wq, const float* __restrict Wk,
                             const float* __restrict Wv, const float* __restrict Wp,
                             __bf16* __restrict Wqt, __bf16* __restrict Wkt,
                             __bf16* __restrict Wvt, __bf16* __restrict Wpt) {
  int i = blockIdx.x * 256 + threadIdx.x;        // i < 768*768 (exact grid)
  int k = i / D_MODEL, n = i % D_MODEL;          // W[k][n] row-major
  int t = n * D_MODEL + k;                       // Wt[n][k]
  Wqt[t] = (__bf16)Wq[i];
  Wkt[t] = (__bf16)Wk[i];
  Wvt[t] = (__bf16)Wv[i];
  Wpt[t] = (__bf16)Wp[i];
}

// ---------------------------------------------------------------------------
// Shared GEMM mainloop: block covers 256 rows x 64 cols; wave w owns 32 rows.
// B tile double-buffered in LDS via async copy; A straight from global (L2).
// Result: acc[2][4] = two 16x64 row-tiles.
// ---------------------------------------------------------------------------
static __device__ __forceinline__ void
gemm_mainloop(const __bf16* __restrict A, const __bf16* __restrict Wt,
              int m0w, int n0, int lane, int tid, __bf16* Bsh, v8f acc[2][4]) {
  int buf = 0;
  stage_b_async(Wt, n0, 0, tid, Bsh);
  wait_async_and_barrier();
  for (int k0 = 0; k0 < D_MODEL; k0 += 32) {
    if (k0 + 32 < D_MODEL)
      stage_b_async(Wt, n0, k0 + 32, tid, Bsh + (buf ^ 1) * (64 * 32));
    v16bf a0 = load_a16x32(A, D_MODEL, m0w,      k0, lane);
    v16bf a1 = load_a16x32(A, D_MODEL, m0w + 16, k0, lane);
    const __bf16* Bcur = Bsh + buf * (64 * 32);
#pragma unroll
    for (int j = 0; j < 4; ++j) {
      v16bf b = load_bT32x16(Bcur, 32, j * 16, 0, lane);
      acc[0][j] = wmma_bf16(a0, b, acc[0][j]);
      acc[1][j] = wmma_bf16(a1, b, acc[1][j]);
    }
    wait_async_and_barrier();   // next buf ready; everyone done with current
    buf ^= 1;
  }
}

// ---------------------------------------------------------------------------
// QKV projection GEMM. vmode 0: store [B,H,S,HD] (Q,K). vmode 1: [B,H,HD,S] (V).
// Grid: (8192/256)*(768/64) = 384 blocks of 256 threads.
// ---------------------------------------------------------------------------
__global__ __launch_bounds__(256)
void qkv_gemm_kernel(const __bf16* __restrict Xb, const __bf16* __restrict Wt,
                     __bf16* __restrict Out, int vmode) {
  __shared__ __bf16 Bsh[2][64 * 32];               // 8KB double buffer
  int tid  = threadIdx.x;
  int lane = tid & 31;
  int w    = tid >> 5;
  int bn = blockIdx.x % (D_MODEL / 64);            // 0..11  (== head)
  int bm = blockIdx.x / (D_MODEL / 64);            // 0..31
  int n0  = bn * 64;
  int m0w = bm * 256 + w * 32;

  const v8f vz = {0.f, 0.f, 0.f, 0.f, 0.f, 0.f, 0.f, 0.f};
  v8f acc[2][4] = {{vz, vz, vz, vz}, {vz, vz, vz, vz}};
  gemm_mainloop(Xb, Wt, m0w, n0, lane, tid, &Bsh[0][0], acc);

  int ncol  = lane & 15;
  int half8 = (lane < 16) ? 0 : 8;
  int h = bn;
#pragma unroll
  for (int rr = 0; rr < 2; ++rr) {
#pragma unroll
    for (int j = 0; j < 4; ++j) {
      int hd = j * 16 + ncol;
#pragma unroll
      for (int r = 0; r < 8; ++r) {
        int row = m0w + rr * 16 + r + half8;       // 0..8191
        int bb = row >> 11, s = row & (SEQ - 1);
        __bf16 v = (__bf16)acc[rr][j][r];
        if (vmode == 0)
          Out[((bb * NHEAD + h) * SEQ + s) * HD + hd] = v;
        else
          Out[((bb * NHEAD + h) * HD + hd) * SEQ + s] = v;
      }
    }
  }
}

// ---------------------------------------------------------------------------
// Flash attention: one wave per (b, h, 16-row q-tile); online softmax over
// 32-wide causal k-tiles; P transposed through LDS to feed the PV WMMA.
// ---------------------------------------------------------------------------
__global__ __launch_bounds__(256)
void attn_kernel(const __bf16* __restrict Qb, const __bf16* __restrict Kb,
                 const __bf16* __restrict Vt, __bf16* __restrict Ob) {
  __shared__ __bf16 psh[8][16 * 32];                  // 1KB per wave

  int lane = threadIdx.x & 31;
  int w    = threadIdx.x >> 5;
  int wid  = blockIdx.x * 8 + w;                      // 6144 waves
  int qt = wid & 127;                                 // S/16 = 128
  int hh = (wid >> 7) % NHEAD;
  int bb = wid / (128 * NHEAD);
  int q0 = qt * 16;

  const __bf16* Qh = Qb + (long)((bb * NHEAD + hh) * SEQ) * HD;
  const __bf16* Kh = Kb + (long)((bb * NHEAD + hh) * SEQ) * HD;
  const __bf16* Vh = Vt + (long)((bb * NHEAD + hh) * HD) * SEQ;
  __bf16* pw = &psh[w][0];

  v16bf aq0 = load_a16x32(Qh, HD, q0, 0, lane);
  v16bf aq1 = load_a16x32(Qh, HD, q0, 32, lane);

  const v8f vz = {0.f, 0.f, 0.f, 0.f, 0.f, 0.f, 0.f, 0.f};
  v8f o0 = vz, o1 = vz, o2 = vz, o3 = vz;
  float mrow[8], lrow[8];
#pragma unroll
  for (int r = 0; r < 8; ++r) { mrow[r] = -1e30f; lrow[r] = 0.f; }

  int ncol  = lane & 15;
  int half8 = (lane < 16) ? 0 : 8;
  const float sc = 0.125f;                            // HD^-0.5

  for (int kb0 = 0; kb0 <= q0 + 15; kb0 += 32) {      // causal tile skipping
    v8f s0 = vz, s1 = vz;
    s0 = wmma_bf16(aq0, load_bT32x16(Kh, HD, kb0,       0,  lane), s0);
    s0 = wmma_bf16(aq1, load_bT32x16(Kh, HD, kb0,       32, lane), s0);
    s1 = wmma_bf16(aq0, load_bT32x16(Kh, HD, kb0 + 16,  0,  lane), s1);
    s1 = wmma_bf16(aq1, load_bT32x16(Kh, HD, kb0 + 16,  32, lane), s1);

    int c0 = kb0 + ncol, c1 = c0 + 16;
    if (kb0 + 31 > q0) {                              // diagonal tile: mask
#pragma unroll
      for (int r = 0; r < 8; ++r) {
        int qr = q0 + r + half8;
        s0[r] = (c0 <= qr) ? s0[r] * sc : -1e30f;
        s1[r] = (c1 <= qr) ? s1[r] * sc : -1e30f;
      }
    } else {
#pragma unroll
      for (int r = 0; r < 8; ++r) { s0[r] *= sc; s1[r] *= sc; }
    }

    float fac[8];
#pragma unroll
    for (int r = 0; r < 8; ++r) {
      float mx = fmaxf(s0[r], s1[r]);
      mx = fmaxf(mx, __shfl_xor(mx, 1, 32));
      mx = fmaxf(mx, __shfl_xor(mx, 2, 32));
      mx = fmaxf(mx, __shfl_xor(mx, 4, 32));
      mx = fmaxf(mx, __shfl_xor(mx, 8, 32));
      float newm = fmaxf(mrow[r], mx);
      fac[r]  = __expf(mrow[r] - newm);
      mrow[r] = newm;
      float p0 = __expf(s0[r] - newm);
      float p1 = __expf(s1[r] - newm);
      float rs = p0 + p1;
      rs += __shfl_xor(rs, 1, 32);
      rs += __shfl_xor(rs, 2, 32);
      rs += __shfl_xor(rs, 4, 32);
      rs += __shfl_xor(rs, 8, 32);
      lrow[r] = lrow[r] * fac[r] + rs;
      pw[(r + half8) * 32 + ncol]      = (__bf16)p0;
      pw[(r + half8) * 32 + 16 + ncol] = (__bf16)p1;
    }
#pragma unroll
    for (int r = 0; r < 8; ++r) {
      o0[r] *= fac[r]; o1[r] *= fac[r]; o2[r] *= fac[r]; o3[r] *= fac[r];
    }

    // same-wave DS ops are in-order; fence compiler + wait for LDS writes
    asm volatile("s_wait_dscnt 0x0" ::: "memory");

    v16bf ap = load_a16x32(pw, 32, 0, 0, lane);       // ds_load transpose
    o0 = wmma_bf16(ap, load_bT32x16(Vh, SEQ, 0,  kb0, lane), o0);
    o1 = wmma_bf16(ap, load_bT32x16(Vh, SEQ, 16, kb0, lane), o1);
    o2 = wmma_bf16(ap, load_bT32x16(Vh, SEQ, 32, kb0, lane), o2);
    o3 = wmma_bf16(ap, load_bT32x16(Vh, SEQ, 48, kb0, lane), o3);
  }

#pragma unroll
  for (int r = 0; r < 8; ++r) {
    float inv = 1.0f / lrow[r];
    long row  = (long)bb * SEQ + q0 + r + half8;
    long base = row * D_MODEL + hh * HD;
    Ob[base +  0 + ncol] = (__bf16)(o0[r] * inv);
    Ob[base + 16 + ncol] = (__bf16)(o1[r] * inv);
    Ob[base + 32 + ncol] = (__bf16)(o2[r] * inv);
    Ob[base + 48 + ncol] = (__bf16)(o3[r] * inv);
  }
}

// ---------------------------------------------------------------------------
// Output projection: out[8192x768] = Ob * Wp + bp   (fp32 epilogue)
// ---------------------------------------------------------------------------
__global__ __launch_bounds__(256)
void out_proj_kernel(const __bf16* __restrict Ob, const __bf16* __restrict Wpt,
                     const float* __restrict bp, float* __restrict out) {
  __shared__ __bf16 Bsh[2][64 * 32];
  int tid  = threadIdx.x;
  int lane = tid & 31;
  int w    = tid >> 5;
  int bn = blockIdx.x % (D_MODEL / 64);
  int bm = blockIdx.x / (D_MODEL / 64);
  int n0  = bn * 64;
  int m0w = bm * 256 + w * 32;

  const v8f vz = {0.f, 0.f, 0.f, 0.f, 0.f, 0.f, 0.f, 0.f};
  v8f acc[2][4] = {{vz, vz, vz, vz}, {vz, vz, vz, vz}};
  gemm_mainloop(Ob, Wpt, m0w, n0, lane, tid, &Bsh[0][0], acc);

  int ncol  = lane & 15;
  int half8 = (lane < 16) ? 0 : 8;
#pragma unroll
  for (int rr = 0; rr < 2; ++rr) {
#pragma unroll
    for (int j = 0; j < 4; ++j) {
      int c = n0 + j * 16 + ncol;
      float bias = bp[c];
#pragma unroll
      for (int r = 0; r < 8; ++r) {
        long row = m0w + rr * 16 + r + half8;
        out[row * D_MODEL + c] = acc[rr][j][r] + bias;
      }
    }
  }
}

// ---------------------------------------------------------------------------
// Host launcher
// ---------------------------------------------------------------------------
extern "C" void kernel_launch(void* const* d_in, const int* in_sizes, int n_in,
                              void* d_out, int out_size, void* d_ws, size_t ws_size,
                              hipStream_t stream) {
  const float* x  = (const float*)d_in[0];
  const float* Wq = (const float*)d_in[1];
  const float* Wk = (const float*)d_in[2];
  const float* Wv = (const float*)d_in[3];
  const float* Wp = (const float*)d_in[4];
  const float* bp = (const float*)d_in[5];

  char* ws = (char*)d_ws;
  size_t off = 0;
  auto carve = [&](size_t bytes) -> void* {
    void* p = ws + off;
    off += (bytes + 65535) & ~(size_t)65535;   // 64KB pad between buffers
    return p;
  };
  __bf16* Xb  = (__bf16*)carve((size_t)MROWS * D_MODEL * 2);
  __bf16* Wqt = (__bf16*)carve((size_t)D_MODEL * D_MODEL * 2);
  __bf16* Wkt = (__bf16*)carve((size_t)D_MODEL * D_MODEL * 2);
  __bf16* Wvt = (__bf16*)carve((size_t)D_MODEL * D_MODEL * 2);
  __bf16* Wpt = (__bf16*)carve((size_t)D_MODEL * D_MODEL * 2);
  __bf16* Qb  = (__bf16*)carve((size_t)MROWS * D_MODEL * 2);
  __bf16* Kb  = (__bf16*)carve((size_t)MROWS * D_MODEL * 2);
  __bf16* Vt  = (__bf16*)carve((size_t)MROWS * D_MODEL * 2);
  __bf16* Ob  = (__bf16*)carve((size_t)MROWS * D_MODEL * 2);

  // 1) precision prep
  cvt_x_kernel<<<(MROWS * D_MODEL) / 256, 256, 0, stream>>>(x, Xb);
  cvt_w_kernel<<<(D_MODEL * D_MODEL) / 256, 256, 0, stream>>>(
      Wq, Wk, Wv, Wp, Wqt, Wkt, Wvt, Wpt);

  // 2) Q/K/V projections: 384 blocks (256 rows x 64 cols each)
  const int gemm_blocks = (MROWS / 256) * (D_MODEL / 64);  // 384
  qkv_gemm_kernel<<<gemm_blocks, 256, 0, stream>>>(Xb, Wqt, Qb, 0);
  qkv_gemm_kernel<<<gemm_blocks, 256, 0, stream>>>(Xb, Wkt, Kb, 0);
  qkv_gemm_kernel<<<gemm_blocks, 256, 0, stream>>>(Xb, Wvt, Vt, 1);

  // 3) causal flash attention: 4*12*128 = 6144 wave-tiles = 768 blocks
  const int attn_blocks = (BATCH * NHEAD * (SEQ / 16)) / 8;   // 768
  attn_kernel<<<attn_blocks, 256, 0, stream>>>(Qb, Kb, Vt, Ob);

  // 4) output projection + bias -> fp32 d_out
  out_proj_kernel<<<gemm_blocks, 256, 0, stream>>>(Ob, Wpt, bp, (float*)d_out);
}
